// GraphSAGE_79035988181424
// MI455X (gfx1250) — matrix operand order repaired and checked by reading, not compiled, not run on hardware
//
#include <hip/hip_runtime.h>
#include <hip/hip_bf16.h>

// ---------------------------------------------------------------------------
// GraphSAGE forward (4x SAGEConv(mean) + BN + ReLU, mean-pool + MLP scorer)
// CDNA5 / gfx1250: wave32, v_wmma_f32_16x16x32_f16 matrix cores.
// ---------------------------------------------------------------------------

typedef __attribute__((ext_vector_type(16))) _Float16 v16h;
typedef __attribute__((ext_vector_type(8)))  _Float16 v8h;
typedef __attribute__((ext_vector_type(8)))  float    v8f;

union F16Frag {
    v16h v;
    v8h  h8[2];
    _Float16 h[16];
};

#define N_NODES 50000
#define N_EDGES 600000
#define MAX_DIM 256
#define BN_EPS  1e-5f

// ---------------------------------------------------------------------------
// elementwise helpers
// ---------------------------------------------------------------------------
__global__ void zero_f32_kernel(float* __restrict__ p, int n) {
    int i = blockIdx.x * blockDim.x + threadIdx.x;
    int stride = gridDim.x * blockDim.x;
    for (; i < n; i += stride) p[i] = 0.0f;
}

__global__ void deg_kernel(const int* __restrict__ dst, float* __restrict__ deg, int E) {
    int i = blockIdx.x * blockDim.x + threadIdx.x;
    int stride = gridDim.x * blockDim.x;
    for (; i < E; i += stride) atomicAdd(&deg[dst[i]], 1.0f);
}

__global__ void invdeg_kernel(float* __restrict__ deg, int n) {
    int i = blockIdx.x * blockDim.x + threadIdx.x;
    int stride = gridDim.x * blockDim.x;
    for (; i < n; i += stride) {
        float d = deg[i];
        deg[i] = 1.0f / (d > 1.0f ? d : 1.0f);
    }
}

__global__ void cvt_f32_to_f16_kernel(const float* __restrict__ src,
                                      _Float16* __restrict__ dst, int n) {
    int i = blockIdx.x * blockDim.x + threadIdx.x;
    int stride = gridDim.x * blockDim.x;
    for (; i < n; i += stride) dst[i] = (_Float16)src[i];
}

// W [din, dout] f32 row-major -> WT [dout, din] f16 (so WMMA B-frag loads are
// contiguous along K per output column)
__global__ void cvt_transpose_w_kernel(const float* __restrict__ W,
                                       _Float16* __restrict__ WT,
                                       int din, int dout) {
    int i = blockIdx.x * blockDim.x + threadIdx.x;
    int stride = gridDim.x * blockDim.x;
    int n = din * dout;
    for (; i < n; i += stride) {
        int k = i / dout;
        int c = i - k * dout;
        WT[(size_t)c * din + k] = (_Float16)W[i];
    }
}

// agg_f16[i] = (f16)(agg_f32[i] * inv_deg[node]);  din is a power of two
__global__ void scale_cvt_agg_kernel(const float* __restrict__ agg,
                                     const float* __restrict__ invdeg,
                                     _Float16* __restrict__ aggH,
                                     int n, int log2din) {
    int i = blockIdx.x * blockDim.x + threadIdx.x;
    int stride = gridDim.x * blockDim.x;
    for (; i < n; i += stride)
        aggH[i] = (_Float16)(agg[i] * invdeg[i >> log2din]);
}

// ---------------------------------------------------------------------------
// edge-parallel mean-aggregation: agg[dst] += h[src]  (f16 reads, f32 atomics;
// feature tables are L2-resident on the 192MB L2)
// ---------------------------------------------------------------------------
__global__ void gather_kernel(const _Float16* __restrict__ h,
                              const int* __restrict__ src,
                              const int* __restrict__ dst,
                              float* __restrict__ agg,
                              int total, int log2chunks, int din) {
    int t = blockIdx.x * blockDim.x + threadIdx.x;
    int stride = gridDim.x * blockDim.x;
    for (; t < total; t += stride) {
        int e = t >> log2chunks;
        int c = (t - (e << log2chunks)) << 3;   // 8 features per work item
        int s = src[e];
        int d = dst[e];
        v8h hv = *(const v8h*)(h + (size_t)s * din + c);
        float* pa = agg + (size_t)d * din + c;
#pragma unroll
        for (int j = 0; j < 8; ++j)
            atomicAdd(pa + j, (float)hv[j]);
    }
}

// ---------------------------------------------------------------------------
// Fused SAGE layer GEMM:
//   out = ReLU( BN( aggH @ Wl + bl + hH @ Wr ) )
// One wave = one 16x16 output tile (M tile from blockIdx.x, N tile from wave id).
// A/H frags: contiguous k = 16*g + s, same as B (two b128 loads / lane).
// B frags from pre-transposed [N,K] f16 weights (contiguous 32B / lane).
// ---------------------------------------------------------------------------
__global__ void sage_gemm_kernel(const _Float16* __restrict__ A,    // aggH [N_NODES, din]
                                 const _Float16* __restrict__ WlT,  // [dout, din]
                                 const _Float16* __restrict__ H,    // hH   [N_NODES, din]
                                 const _Float16* __restrict__ WrT,  // [dout, din]
                                 const float* __restrict__ bl,
                                 const float* __restrict__ gamma,
                                 const float* __restrict__ beta,
                                 const float* __restrict__ mean,
                                 const float* __restrict__ var,
                                 float* __restrict__ outF32,        // may be null
                                 _Float16* __restrict__ outF16,     // may be null
                                 int din, int dout) {
    const int wave = threadIdx.x >> 5;
    const int lane = threadIdx.x & 31;
    const int g    = lane >> 4;       // lane group 0/1
    const int l15  = lane & 15;
    const int m0   = blockIdx.x * 16;
    const int n0   = wave * 16;
    const int row  = m0 + l15;        // A-frag row for this lane
    const int col  = n0 + l15;        // B-frag / output column for this lane

    v8f acc = {};

    // agg @ Wl
    for (int k = 0; k < din; k += 32) {
        F16Frag a, b;
        const _Float16* pa = A + (size_t)row * din + k + g * 16;
        a.h8[0] = *(const v8h*)(pa);
        a.h8[1] = *(const v8h*)(pa + 8);
        const _Float16* pb = WlT + (size_t)col * din + k + g * 16;
        b.v = *(const v16h*)(pb);
        acc = __builtin_amdgcn_wmma_f32_16x16x32_f16(
            false, a.v, false, b.v, (short)0, acc, false, false);
    }
    // h @ Wr
    for (int k = 0; k < din; k += 32) {
        F16Frag a, b;
        const _Float16* pa = H + (size_t)row * din + k + g * 16;
        a.h8[0] = *(const v8h*)(pa);
        a.h8[1] = *(const v8h*)(pa + 8);
        const _Float16* pb = WrT + (size_t)col * din + k + g * 16;
        b.v = *(const v16h*)(pb);
        acc = __builtin_amdgcn_wmma_f32_16x16x32_f16(
            false, a.v, false, b.v, (short)0, acc, false, false);
    }

    // fused bias + BatchNorm(eval) + ReLU epilogue
    const float bv = bl[col];
    const float mu = mean[col];
    const float sc = rsqrtf(var[col] + BN_EPS) * gamma[col];
    const float be = beta[col];
#pragma unroll
    for (int r = 0; r < 8; ++r) {
        int m = m0 + r + g * 8;               // C/D layout: VGPR r -> M = r + 8*g
        float v = acc[r] + bv;
        v = (v - mu) * sc + be;
        v = v > 0.0f ? v : 0.0f;
        size_t o = (size_t)m * dout + col;
        if (outF32) outF32[o] = v;
        if (outF16) outF16[o] = (_Float16)v;
    }
}

// ---------------------------------------------------------------------------
// graph embedding = column mean of final h [N_NODES, 32]
// ---------------------------------------------------------------------------
__global__ void colmean_kernel(const float* __restrict__ h,
                               float* __restrict__ gmean,
                               int nnodes, int dout) {
    __shared__ float sm[256];
    int c = blockIdx.x;
    float s = 0.0f;
    for (int n = threadIdx.x; n < nnodes; n += blockDim.x)
        s += h[(size_t)n * dout + c];
    sm[threadIdx.x] = s;
    __syncthreads();
    for (int off = 128; off > 0; off >>= 1) {
        if ((int)threadIdx.x < off) sm[threadIdx.x] += sm[threadIdx.x + off];
        __syncthreads();
    }
    if (threadIdx.x == 0) gmean[c] = sm[0] / (float)nnodes;
}

// score = sigmoid( relu(g @ W1 + b1) @ W2 + b2 )   (32 -> 16 -> 1)
__global__ void scorer_kernel(const float* __restrict__ g,
                              const float* __restrict__ W1,
                              const float* __restrict__ b1,
                              const float* __restrict__ W2,
                              const float* __restrict__ b2,
                              float* __restrict__ out) {
    if (blockIdx.x == 0 && threadIdx.x == 0) {
        float s = b2[0];
        for (int j = 0; j < 16; ++j) {
            float hj = b1[j];
            for (int c = 0; c < 32; ++c) hj += g[c] * W1[c * 16 + j];
            hj = hj > 0.0f ? hj : 0.0f;
            s += hj * W2[j];
        }
        out[0] = 1.0f / (1.0f + __expf(-s));
    }
}

// ---------------------------------------------------------------------------
static inline int ilog2i(int v) { int l = 0; while ((1 << l) < v) ++l; return l; }

extern "C" void kernel_launch(void* const* d_in, const int* in_sizes, int n_in,
                              void* d_out, int out_size, void* d_ws, size_t ws_size,
                              hipStream_t stream) {
    (void)in_sizes; (void)n_in; (void)out_size; (void)ws_size;

    const float* x    = (const float*)d_in[0];
    const int*   eidx = (const int*)d_in[1];      // [2, E]: row 0 = src, row 1 = dst
    const int*   esrc = eidx;
    const int*   edst = eidx + N_EDGES;

    const int dims[5] = {128, 256, 128, 64, 32};

    struct Layer { const float *Wl, *bl, *Wr, *gamma, *beta, *mean, *var; };
    Layer L[4];
    for (int i = 0; i < 4; ++i) {
        int b = 2 + 7 * i;
        L[i].Wl    = (const float*)d_in[b + 0];
        L[i].bl    = (const float*)d_in[b + 1];
        L[i].Wr    = (const float*)d_in[b + 2];
        L[i].gamma = (const float*)d_in[b + 3];
        L[i].beta  = (const float*)d_in[b + 4];
        L[i].mean  = (const float*)d_in[b + 5];
        L[i].var   = (const float*)d_in[b + 6];
    }
    const float* sW1 = (const float*)d_in[30];
    const float* sb1 = (const float*)d_in[31];
    const float* sW2 = (const float*)d_in[32];
    const float* sb2 = (const float*)d_in[33];

    // ---- workspace carving (all 256B aligned) ----
    char* ws = (char*)d_ws;
    size_t off = 0;
    auto carve = [&](size_t bytes) -> char* {
        char* p = ws + off;
        off += (bytes + 255) & ~(size_t)255;
        return p;
    };
    float*    invdeg = (float*)   carve((size_t)N_NODES * 4);
    float*    agg    = (float*)   carve((size_t)N_NODES * MAX_DIM * 4);
    _Float16* aggH   = (_Float16*)carve((size_t)N_NODES * MAX_DIM * 2);
    _Float16* hA     = (_Float16*)carve((size_t)N_NODES * MAX_DIM * 2);
    _Float16* hB     = (_Float16*)carve((size_t)N_NODES * MAX_DIM * 2);
    _Float16* wlT    = (_Float16*)carve((size_t)MAX_DIM * MAX_DIM * 2);
    _Float16* wrT    = (_Float16*)carve((size_t)MAX_DIM * MAX_DIM * 2);
    float*    gmean  = (float*)   carve(32 * 4);

    float* outNodes = (float*)d_out;            // [N_NODES, 32]
    float* outScore = outNodes + (size_t)N_NODES * 32;

    const int TB = 256;
    const int EW_GRID = 4096;                   // grid-stride elementwise grid

    // ---- degree -> inv_deg ----
    zero_f32_kernel<<<EW_GRID, TB, 0, stream>>>(invdeg, N_NODES);
    deg_kernel<<<EW_GRID, TB, 0, stream>>>(edst, invdeg, N_EDGES);
    invdeg_kernel<<<EW_GRID, TB, 0, stream>>>(invdeg, N_NODES);

    // ---- x -> f16 ----
    cvt_f32_to_f16_kernel<<<EW_GRID, TB, 0, stream>>>(x, hA, N_NODES * 128);

    _Float16* h16 = hA;
    _Float16* nxt = hB;
    for (int l = 0; l < 4; ++l) {
        const int din  = dims[l];
        const int dout = dims[l + 1];
        const int log2din = ilog2i(din);
        const bool last = (l == 3);

        // 1) zero agg
        zero_f32_kernel<<<EW_GRID, TB, 0, stream>>>(agg, N_NODES * din);

        // 2) edge gather + scatter-add (8 features per work item)
        int chunks = din >> 3;
        int total  = N_EDGES * chunks;
        gather_kernel<<<EW_GRID, TB, 0, stream>>>(h16, esrc, edst, agg,
                                                  total, ilog2i(chunks), din);

        // 3) mean-scale + f16 convert
        scale_cvt_agg_kernel<<<EW_GRID, TB, 0, stream>>>(agg, invdeg, aggH,
                                                         N_NODES * din, log2din);

        // 4) weight transpose + f16 convert
        int wn = din * dout;
        cvt_transpose_w_kernel<<<(wn + TB - 1) / TB, TB, 0, stream>>>(L[l].Wl, wlT, din, dout);
        cvt_transpose_w_kernel<<<(wn + TB - 1) / TB, TB, 0, stream>>>(L[l].Wr, wrT, din, dout);

        // 5) fused WMMA GEMM + bias + BN + ReLU
        dim3 grid(N_NODES / 16);                 // 3125 M tiles (exact)
        dim3 block((dout / 16) * 32);            // one wave per 16-wide N tile
        sage_gemm_kernel<<<grid, block, 0, stream>>>(
            aggH, wlT, h16, wrT,
            L[l].bl, L[l].gamma, L[l].beta, L[l].mean, L[l].var,
            last ? outNodes : nullptr,
            last ? nullptr  : nxt,
            din, dout);

        _Float16* t = h16; h16 = nxt; nxt = t;
    }

    // ---- graph embedding + scorer ----
    colmean_kernel<<<32, 256, 0, stream>>>(outNodes, gmean, N_NODES, 32);
    scorer_kernel<<<1, 32, 0, stream>>>(gmean, sW1, sb1, sW2, sb2, outScore);
}
